// FCOS_72430328480136
// MI455X (gfx1250) — compile-verified, matrix operand order checked by reading, not run on hardware
//
#include <hip/hip_runtime.h>
#include <hip/hip_bf16.h>
#include <stdint.h>

typedef __attribute__((ext_vector_type(16))) _Float16 v16h;
typedef __attribute__((ext_vector_type(8)))  _Float16 v8h;
typedef __attribute__((ext_vector_type(8)))  float    v8f;
typedef __attribute__((ext_vector_type(4)))  int      v4i;
typedef __attribute__((ext_vector_type(8)))  int      v8i;
typedef __attribute__((ext_vector_type(4)))  unsigned int v4u;

#define KTOT 2304
#define SA_STR 40
#define SB_STR 40

#if defined(__has_builtin)
#if __has_builtin(__builtin_amdgcn_global_load_async_to_lds_b128) && \
    __has_builtin(__builtin_amdgcn_s_wait_asynccnt)
#define USE_ASYNC 1
#endif
#endif

#if defined(USE_ASYNC)
#if __has_builtin(__builtin_amdgcn_tensor_load_to_lds) && \
    __has_builtin(__builtin_amdgcn_s_wait_tensorcnt)
#define USE_TDM 1
#endif
#endif

#ifdef USE_ASYNC
// builtin signature (from hipcc diagnostic): (v4i* src_global, v4i* dst_lds,
// imm offset, imm cpol); LLVM infers AS1/AS3 from pointer provenance.
__device__ __forceinline__ void async_ld16(const void* g, void* l) {
  __builtin_amdgcn_global_load_async_to_lds_b128((v4i*)g, (v4i*)l, 0, 0);
}
#endif

#ifdef USE_TDM
// One TDM descriptor: 2-D tile of 128 rows x 64B from global (row stride
// KTOT*2 B) into LDS with 16B padding after every 64B (-> 80B LDS row stride).
// This toolchain exposes the 6-arg form: (g0, g1, g2, g3, g_extra, cpol).
__device__ __forceinline__ void tdm_load_b(const void* g, void* l) {
  unsigned long long ga = (unsigned long long)(uintptr_t)g;
  unsigned lds = (unsigned)(uintptr_t)l;   // low 32 bits of generic LDS addr
  v4u g0 = { 1u,                                   // count=1, user mode
             lds,                                  // lds_addr (bytes)
             (unsigned)(ga & 0xffffffffu),         // global_addr[31:0]
             (unsigned)(((ga >> 32) & 0x1ffffffu)  // global_addr[56:32]
                        | (2u << 30)) };           // type = 2 (image)
  v8i g1 = { (int)((1u << 16)      // data_size = 2 bytes
                 | (1u << 20)      // pad_enable
                 | (3u << 22)      // pad_interval: 16 dwords (64B)
                 | (3u << 25)),    // pad_amount : 4 dwords (16B)
             (int)(((unsigned)KTOT & 0xffffu) << 16), // tensor_dim0 lo16
             (int)(0x4000u << 16), // tensor_dim0 hi16=0 | tensor_dim1 lo16
             (int)(32u << 16),     // tensor_dim1 hi16=0 | tile_dim0 = 32 elems
             (int)128u,            // tile_dim1 = 128 rows | tile_dim2 = 0
             (int)KTOT,            // tensor_dim0_stride[31:0] (elements)
             0,                    // stride0 hi | tensor_dim1_stride lo
             0 };
  v4i gz4 = { 0, 0, 0, 0 };
  v8i gz8 = { 0, 0, 0, 0, 0, 0, 0, 0 };
  __builtin_amdgcn_tensor_load_to_lds(g0, g1, gz4, gz4, gz8, 0);
}
#endif

// ---------------------------------------------------------------------------
// Implicit-GEMM 3x3 conv (stride 1, SAME) via WMMA f32_16x16x32_f16 with a
// double-buffered LDS pipeline: TDM for weight tiles + async global->LDS for
// the im2col gather (falls back to sync staging if builtins are absent).
// act : NHWC f16  [Mtotal][256]   (Mtotal = 2*H*W)
// wt  : n-major f16 [Npad][2304], k = tap*256 + c, tap = ky*3+kx (rows >= Nout zero)
// out : NHWC, f16 (towers) or f32 (heads), row stride Nstride
// zp  : 512B zero-filled global page (im2col border source)
// ---------------------------------------------------------------------------
__global__ __launch_bounds__(256) void conv3x3_wmma(
    const _Float16* __restrict__ act, const _Float16* __restrict__ wt,
    void* __restrict__ outp, const float* __restrict__ bias,
    const _Float16* __restrict__ zp,
    int H, int W, int Mtotal, int Npad, int Nstride, int Nreal,
    int relu, int out32)
{
  __shared__ _Float16 sA[2][64 * SA_STR];
  __shared__ _Float16 sB[2][128 * SB_STR];

  const int tid   = threadIdx.x;
  const int lane  = tid & 31;
  const int wave  = tid >> 5;
  const int mtile = wave & 3;   // M offset 16*mtile
  const int ngrp  = wave >> 2;  // N offset 64*ngrp
  const int mbase = blockIdx.x * 64;
  const int nbase = blockIdx.y * 128;

  // A staging role: 4 threads per location row, 8 channels each
  const int arow = tid >> 2;           // 0..63
  const int acq  = (tid & 3) << 3;     // 0,8,16,24
  const int loc  = mbase + arow;
  const int HWl  = H * W;
  const bool inb = loc < Mtotal;
  int bidx = 0, ay = 0, ax = 0;
  if (inb) { bidx = loc / HWl; int r = loc - bidx * HWl; ay = r / W; ax = r - ay * W; }

  // B staging role: 2 threads per out-channel row, 16 k each (always valid)
  const int brow = tid >> 1;           // 0..127
  const int bkq  = (tid & 1) << 4;     // 0,16
  const _Float16* wrow = wt + (size_t)(nbase + brow) * KTOT;
  const _Float16* wtile = wt + (size_t)nbase * KTOT;   // TDM tile base

  // fragment addressing (ISA 16-bit A/B layout)
  const int la = lane & 15;
  const int kb = (lane >> 4) << 3;     // 0 (lanes 0-15) or 8 (lanes 16-31)

  v8f acc0 = {0.f,0.f,0.f,0.f,0.f,0.f,0.f,0.f};
  v8f acc1 = acc0, acc2 = acc0, acc3 = acc0;

  // incremental tap state for the NEXT issue (no div/mod in the loop)
  int ncq = 0, ndx = -1, ndy = -1;
  const _Float16* aBase = zp;
  auto refreshTap = [&]() {
    const int yy = ay + ndy, xx = ax + ndx;
    const bool av = inb && (yy >= 0) && (yy < H) && (xx >= 0) && (xx < W);
    aBase = av ? (act + ((size_t)((bidx * H + yy) * W + xx)) * 256 + acq) : zp;
  };
  auto advance = [&]() {
    if (++ncq == 8) { ncq = 0; if (++ndx > 1) { ndx = -1; ++ndy; } refreshTap(); }
  };

  auto issue = [&](int s, int buf) {
    const _Float16* gA = aBase + (ncq << 5);          // zero page covers 512B
    _Float16* dA = &sA[buf][arow * SA_STR + acq];
#ifdef USE_ASYNC
    async_ld16(gA, dA);
# ifdef USE_TDM
    if (wave == 0) tdm_load_b(wtile + ((size_t)s << 5), &sB[buf][0]);
# else
    const _Float16* gB = wrow + (s << 5) + bkq;
    _Float16* dB = &sB[buf][brow * SB_STR + bkq];
    async_ld16(gB, dB);
    async_ld16(gB + 8, dB + 8);
# endif
#else
    const _Float16* gB = wrow + (s << 5) + bkq;
    _Float16* dB = &sB[buf][brow * SB_STR + bkq];
    v8h va  = *(const v8h*)gA;
    v8h vb0 = *(const v8h*)gB;
    v8h vb1 = *(const v8h*)(gB + 8);
    *(v8h*)dA       = va;
    *(v8h*)dB       = vb0;
    *(v8h*)(dB + 8) = vb1;
#endif
  };

  auto compute = [&](int buf) {
    const _Float16* pA = &sA[buf][(mtile * 16 + la) * SA_STR + kb];
    v8h a0 = *(const v8h*)pA;
    v8h a1 = *(const v8h*)(pA + 16);
    v16h afr = __builtin_shufflevector(a0, a1,
                 0,1,2,3,4,5,6,7,8,9,10,11,12,13,14,15);
#define BFRAG(NT, ACC)                                                         \
    { const _Float16* pB = &sB[buf][(ngrp * 64 + (NT) * 16 + la) * SB_STR + kb];\
      v8h b0 = *(const v8h*)pB;                                                \
      v8h b1 = *(const v8h*)(pB + 16);                                         \
      v16h bfr = __builtin_shufflevector(b0, b1,                               \
                   0,1,2,3,4,5,6,7,8,9,10,11,12,13,14,15);                     \
      ACC = __builtin_amdgcn_wmma_f32_16x16x32_f16(                            \
               false, afr, false, bfr, (short)0, ACC, false, false); }
    BFRAG(0, acc0)
    BFRAG(1, acc1)
    BFRAG(2, acc2)
    BFRAG(3, acc3)
#undef BFRAG
  };

  // 72 k-steps (9 taps x 8 channel chunks), double-buffered pipeline
  refreshTap();
  issue(0, 0);
#pragma unroll 1
  for (int s = 0; s < 71; ++s) {
    advance();
    issue(s + 1, (s + 1) & 1);
#ifdef USE_ASYNC
# ifdef USE_TDM
    __builtin_amdgcn_s_wait_asynccnt(1);   // tile-s A gather retired
    __builtin_amdgcn_s_wait_tensorcnt(1);  // tile-s B TDM retired (wave 0)
# else
    __builtin_amdgcn_s_wait_asynccnt(3);   // tile-s loads retired (in-order)
# endif
#endif
    __syncthreads();
    compute(s & 1);
    __syncthreads();
  }
#ifdef USE_ASYNC
  __builtin_amdgcn_s_wait_asynccnt(0);
# ifdef USE_TDM
  __builtin_amdgcn_s_wait_tensorcnt(0);
# endif
#endif
  __syncthreads();
  compute(1);

  // epilogue: C/D layout -> NHWC store with bias (+ReLU)
  const int mrow0 = mbase + mtile * 16 + ((lane >> 4) << 3);
  const int ncl   = nbase + ngrp * 64 + la;
  float*    of = (float*)outp;
  _Float16* oh = (_Float16*)outp;
#pragma unroll
  for (int nt = 0; nt < 4; ++nt) {
    const int n = ncl + nt * 16;
    v8f a = (nt == 0) ? acc0 : (nt == 1) ? acc1 : (nt == 2) ? acc2 : acc3;
    if (n < Nreal) {
      const float bs = bias[n];
#pragma unroll
      for (int r = 0; r < 8; ++r) {
        const int m = mrow0 + r;
        if (m < Mtotal) {
          float v = a[r] + bs;
          if (relu) v = fmaxf(v, 0.f);
          if (out32) of[(size_t)m * Nstride + n] = v;
          else       oh[(size_t)m * Nstride + n] = (_Float16)v;
        }
      }
    }
  }
}

// ---------------------------------------------------------------------------
// Weight packing: w[O][256][3][3] fp32 -> dst[Npad][2304] f16 (n-major)
// ---------------------------------------------------------------------------
__global__ void pack_weights(const float* __restrict__ w, _Float16* __restrict__ dst,
                             int Nout, int Npad)
{
  int idx = blockIdx.x * 256 + threadIdx.x;
  int tot = Npad * KTOT;
  if (idx >= tot) return;
  int o = idx / KTOT, k = idx - o * KTOT;
  int tap = k >> 8, c = k & 255;
  float v = (o < Nout) ? w[((size_t)(o * 256 + c)) * 9 + tap] : 0.f;
  dst[idx] = (_Float16)v;
}

// NCHW fp32 -> NHWC f16
__global__ void nchw_to_nhwc(const float* __restrict__ f, _Float16* __restrict__ o,
                             int H, int W)
{
  int idx = blockIdx.x * 256 + threadIdx.x;
  int HW = H * W;
  int tot = 2 * HW * 256;
  if (idx >= tot) return;
  int loc = idx >> 8, c = idx & 255;
  int b = loc / HW, r = loc - b * HW;
  o[idx] = (_Float16)f[((size_t)(b * 256 + c)) * HW + r];
}

__global__ void zero_i32(int* p, int n)
{
  int i = blockIdx.x * 128 + threadIdx.x;
  if (i < n) p[i] = 0;
}

__device__ __forceinline__ float sigf(float x) { return 1.f / (1.f + expf(-x)); }

// score = sqrt(sigmoid(logit)*sigmoid(ctr)); keep > 0.05 as candidates
__global__ void score_compact(const float* __restrict__ logits,
                              const float* __restrict__ ctrO,
                              int HW, int lvl,
                              float* __restrict__ candV, int* __restrict__ candI,
                              int* __restrict__ cnt)
{
  int idx = blockIdx.x * 256 + threadIdx.x;
  int tot = 2 * HW * 80;
  if (idx >= tot) return;
  int loc = idx / 80, c = idx - loc * 80;
  int b = loc / HW, li = loc - b * HW;
  float s = sqrtf(sigf(logits[(size_t)loc * 80 + c]) * sigf(ctrO[(size_t)loc * 16]));
  if (s > 0.05f) {
    int sl = b * 5 + lvl;
    int p = atomicAdd(&cnt[sl], 1);
    if (p < 4096) { candV[sl * 4096 + p] = s; candI[sl * 4096 + p] = li * 80 + c; }
  }
}

// per (batch,level): top-1000 by repeated block-argmax over candidates
__global__ void select_topk(const float* __restrict__ candV, const int* __restrict__ candI,
                            const int* __restrict__ cnt, int lvl,
                            float* __restrict__ selV, int* __restrict__ selI)
{
  int b = blockIdx.x, tid = threadIdx.x;
  int sl = b * 5 + lvl;
  int n = cnt[sl]; if (n > 4096) n = 4096;
  const float* cv = candV + sl * 4096;
  const int*   ci = candI + sl * 4096;
  __shared__ unsigned char used[4096];
  __shared__ float red[256];
  __shared__ int   redi[256];
  for (int i = tid; i < 4096; i += 256) used[i] = 0;
  __syncthreads();
  for (int j = 0; j < 1000; ++j) {
    float bv = 0.f; int bi = -1;
    for (int t = tid; t < n; t += 256)
      if (!used[t] && cv[t] > bv) { bv = cv[t]; bi = t; }
    red[tid] = bv; redi[tid] = bi;
    __syncthreads();
    for (int s = 128; s > 0; s >>= 1) {
      if (tid < s && red[tid + s] > red[tid]) { red[tid] = red[tid + s]; redi[tid] = redi[tid + s]; }
      __syncthreads();
    }
    if (tid == 0) {
      int wsel = redi[0];
      if (wsel >= 0) { used[wsel] = 1; selV[b * 1000 + j] = red[0]; selI[b * 1000 + j] = ci[wsel]; }
      else           { selV[b * 1000 + j] = 0.f; selI[b * 1000 + j] = -1; }
    }
    __syncthreads();
  }
}

// anchor decode + clip, write into per-batch concatenated arrays
__global__ void decode_level(const float* __restrict__ selV, const int* __restrict__ selI,
                             const float* __restrict__ deltas,
                             int lvl, int H, int W, float strd,
                             float* __restrict__ catB, float* __restrict__ catS,
                             int* __restrict__ catC)
{
  int idx = blockIdx.x * 256 + threadIdx.x;
  if (idx >= 2000) return;
  int b = idx / 1000, j = idx - b * 1000;
  float sc = selV[b * 1000 + j];
  int id = selI[b * 1000 + j];
  int pos = b * 5000 + lvl * 1000 + j;
  float x1 = 0.f, y1 = 0.f, x2 = 0.f, y2 = 0.f; int c = 0;
  if (id >= 0) {
    int a = id / 80; c = id - a * 80;
    const float* d = &deltas[(size_t)(b * H * W + a) * 16];
    float ddx = d[0], ddy = d[1], dw = d[2], dh = d[3];
    int ayi = a / W, axi = a - ayi * W;
    float cx = (axi + 0.5f) * strd, cy = (ayi + 0.5f) * strd;
    float wa = 4.f * strd;
    const float SC = 4.135166556742356f;  // log(1000/16)
    dw = fminf(dw, SC); dh = fminf(dh, SC);
    float ncx = ddx * wa + cx, ncy = ddy * wa + cy;
    float nw = expf(dw) * wa, nh = expf(dh) * wa;
    x1 = fminf(fmaxf(ncx - 0.5f * nw, 0.f), 1344.f);
    y1 = fminf(fmaxf(ncy - 0.5f * nh, 0.f), 1024.f);
    x2 = fminf(fmaxf(ncx + 0.5f * nw, 0.f), 1344.f);
    y2 = fminf(fmaxf(ncy + 0.5f * nh, 0.f), 1024.f);
  }
  catB[(size_t)pos * 4 + 0] = x1; catB[(size_t)pos * 4 + 1] = y1;
  catB[(size_t)pos * 4 + 2] = x2; catB[(size_t)pos * 4 + 3] = y2;
  catS[pos] = sc; catC[pos] = c;
}

// class-aware greedy NMS over 5000 entries per batch, emit top 100
__global__ void nms_kernel(const float* __restrict__ catB, const float* __restrict__ catS,
                           const int* __restrict__ catC, float* __restrict__ outp)
{
  const int M = 5000;
  int b = blockIdx.x, tid = threadIdx.x;
  __shared__ float sc[5000];
  __shared__ int   ord[5000];
  __shared__ unsigned char keep[5000];
  __shared__ float red[256];
  __shared__ int   redi[256];
  __shared__ float bi4[5];
  __shared__ int   alive;

  for (int i = tid; i < M; i += 256) { sc[i] = catS[b * 5000 + i]; keep[i] = 0; }
  __syncthreads();

  // argsort descending (keep[] used as "taken" flag during sort)
  for (int j = 0; j < M; ++j) {
    float bv = -1e30f; int bi = -1;
    for (int t = tid; t < M; t += 256)
      if (!keep[t] && sc[t] > bv) { bv = sc[t]; bi = t; }
    red[tid] = bv; redi[tid] = bi;
    __syncthreads();
    for (int s = 128; s > 0; s >>= 1) {
      if (tid < s && red[tid + s] > red[tid]) { red[tid] = red[tid + s]; redi[tid] = redi[tid + s]; }
      __syncthreads();
    }
    if (tid == 0) { ord[j] = redi[0]; keep[redi[0]] = 1; }
    __syncthreads();
  }
  for (int i = tid; i < M; i += 256) keep[i] = 1;
  __syncthreads();

  for (int i = 0; i < M; ++i) {
    if (tid == 0) {
      int o = ord[i];
      alive = (keep[i] && sc[o] > 0.f) ? 1 : 0;
      if (alive) {
        float off = 2000.f * (float)catC[b * 5000 + o];
        float xx1 = catB[(size_t)(b * 5000 + o) * 4 + 0] + off;
        float yy1 = catB[(size_t)(b * 5000 + o) * 4 + 1] + off;
        float xx2 = catB[(size_t)(b * 5000 + o) * 4 + 2] + off;
        float yy2 = catB[(size_t)(b * 5000 + o) * 4 + 3] + off;
        bi4[0] = xx1; bi4[1] = yy1; bi4[2] = xx2; bi4[3] = yy2;
        bi4[4] = (xx2 - xx1) * (yy2 - yy1);
      }
    }
    __syncthreads();
    if (alive) {
      for (int jj = i + 1 + tid; jj < M; jj += 256) {
        if (keep[jj]) {
          int o2 = ord[jj];
          float off = 2000.f * (float)catC[b * 5000 + o2];
          float x1 = catB[(size_t)(b * 5000 + o2) * 4 + 0] + off;
          float y1 = catB[(size_t)(b * 5000 + o2) * 4 + 1] + off;
          float x2 = catB[(size_t)(b * 5000 + o2) * 4 + 2] + off;
          float y2 = catB[(size_t)(b * 5000 + o2) * 4 + 3] + off;
          float iw = fminf(bi4[2], x2) - fmaxf(bi4[0], x1);
          float ih = fminf(bi4[3], y2) - fmaxf(bi4[1], y1);
          float inter = fmaxf(iw, 0.f) * fmaxf(ih, 0.f);
          float areaj = (x2 - x1) * (y2 - y1);
          float iou = inter / (bi4[4] + areaj - inter + 1e-9f);
          if (iou > 0.6f) keep[jj] = 0;
        }
      }
    }
    __syncthreads();
  }

  if (tid == 0) {
    int c = 0;
    for (int i = 0; i < M && c < 100; ++i) {
      int o = ord[i];
      if (keep[i] && sc[o] > 0.f) {
        for (int k = 0; k < 4; ++k)
          outp[(size_t)(b * 100 + c) * 4 + k] = catB[(size_t)(b * 5000 + o) * 4 + k];
        outp[800 + b * 100 + c]  = sc[o];
        outp[1000 + b * 100 + c] = (float)catC[b * 5000 + o];
        ++c;
      }
    }
    for (; c < 100; ++c) {
      for (int k = 0; k < 4; ++k) outp[(size_t)(b * 100 + c) * 4 + k] = 0.f;
      outp[800 + b * 100 + c]  = 0.f;
      outp[1000 + b * 100 + c] = 0.f;
    }
  }
}

// ---------------------------------------------------------------------------
extern "C" void kernel_launch(void* const* d_in, const int* in_sizes, int n_in,
                              void* d_out, int out_size, void* d_ws, size_t ws_size,
                              hipStream_t stream) {
  const float* cls_w   = (const float*)d_in[5];
  const float* cls_b   = (const float*)d_in[6];
  const float* box_w   = (const float*)d_in[7];
  const float* box_b   = (const float*)d_in[8];
  const float* score_w = (const float*)d_in[9];
  const float* score_b = (const float*)d_in[10];
  const float* pred_w  = (const float*)d_in[11];
  const float* pred_b  = (const float*)d_in[12];
  const float* ctr_w   = (const float*)d_in[13];
  const float* ctr_b   = (const float*)d_in[14];

  char* wsb = (char*)d_ws;
  size_t cur = 0;
  auto take = [&](size_t bytes) -> void* {
    void* p = wsb + cur;
    cur += (bytes + 255) & ~(size_t)255;
    return p;
  };

  const size_t WSTRIDE = (size_t)256 * KTOT;          // halfs per tower conv
  _Float16* wtCls   = (_Float16*)take(4 * WSTRIDE * 2);
  _Float16* wtBox   = (_Float16*)take(4 * WSTRIDE * 2);
  _Float16* wtScore = (_Float16*)take((size_t)128 * KTOT * 2);
  _Float16* wtPred  = (_Float16*)take((size_t)128 * KTOT * 2);
  _Float16* wtCtr   = (_Float16*)take((size_t)128 * KTOT * 2);

  const size_t MAXLOC = (size_t)2 * 128 * 168;        // 43008 locations max
  _Float16* featC = (_Float16*)take(MAXLOC * 256 * 2);
  _Float16* actA  = (_Float16*)take(MAXLOC * 256 * 2);
  _Float16* actB  = (_Float16*)take(MAXLOC * 256 * 2);
  float* logits = (float*)take(MAXLOC * 80 * 4);
  float* ctrO   = (float*)take(MAXLOC * 16 * 4);
  float* deltas = (float*)take(MAXLOC * 16 * 4);
  float* candV  = (float*)take((size_t)10 * 4096 * 4);
  int*   candI  = (int*)  take((size_t)10 * 4096 * 4);
  int*   cnt    = (int*)  take(64);
  float* selV   = (float*)take(2000 * 4);
  int*   selI   = (int*)  take(2000 * 4);
  float* catB   = (float*)take((size_t)2 * 5000 * 4 * 4);
  float* catS   = (float*)take((size_t)2 * 5000 * 4);
  int*   catC   = (int*)  take((size_t)2 * 5000 * 4);
  _Float16* zpage = (_Float16*)take(512);             // zero page for borders

  // pack weights to f16, n-major [Npad][2304], rows >= Nout zeroed
  auto pack = [&](const float* w, _Float16* dst, int Nout, int Npad) {
    int tot = Npad * KTOT;
    pack_weights<<<(tot + 255) / 256, 256, 0, stream>>>(w, dst, Nout, Npad);
  };
  for (int i = 0; i < 4; ++i) {
    pack(cls_w + (size_t)i * WSTRIDE, wtCls + (size_t)i * WSTRIDE, 256, 256);
    pack(box_w + (size_t)i * WSTRIDE, wtBox + (size_t)i * WSTRIDE, 256, 256);
  }
  pack(score_w, wtScore, 80, 128);
  pack(pred_w,  wtPred,   4, 128);
  pack(ctr_w,   wtCtr,    1, 128);

  zero_i32<<<1, 128, 0, stream>>>(cnt, 10);
  zero_i32<<<1, 128, 0, stream>>>((int*)zpage, 128);

  auto conv = [&](const _Float16* in, const _Float16* wtp, const float* bias,
                  void* outp, int H, int W, int Npad, int Nstride, int Nreal,
                  int relu, int out32) {
    int Mtot = 2 * H * W;
    dim3 g((Mtot + 63) / 64, Npad / 128);
    conv3x3_wmma<<<g, 256, 0, stream>>>(in, wtp, outp, bias, zpage, H, W, Mtot,
                                        Npad, Nstride, Nreal, relu, out32);
  };

  const int LH[5] = {128, 64, 32, 16, 8};
  const int LW[5] = {168, 84, 42, 21, 11};
  const float LS[5] = {8.f, 16.f, 32.f, 64.f, 128.f};

  for (int l = 0; l < 5; ++l) {
    const int H = LH[l], W = LW[l], HW = H * W;
    const float* feat = (const float*)d_in[l];
    nchw_to_nhwc<<<(2 * HW * 256 + 255) / 256, 256, 0, stream>>>(feat, featC, H, W);

    // cls tower
    conv(featC, wtCls + 0 * WSTRIDE, cls_b + 0,   actA, H, W, 256, 256, 256, 1, 0);
    conv(actA,  wtCls + 1 * WSTRIDE, cls_b + 256, actB, H, W, 256, 256, 256, 1, 0);
    conv(actB,  wtCls + 2 * WSTRIDE, cls_b + 512, actA, H, W, 256, 256, 256, 1, 0);
    conv(actA,  wtCls + 3 * WSTRIDE, cls_b + 768, actB, H, W, 256, 256, 256, 1, 0);
    conv(actB,  wtScore, score_b, logits, H, W, 128, 80, 80, 0, 1);
    conv(actB,  wtCtr,   ctr_b,   ctrO,   H, W, 128, 16,  1, 0, 1);

    // box tower
    conv(featC, wtBox + 0 * WSTRIDE, box_b + 0,   actA, H, W, 256, 256, 256, 1, 0);
    conv(actA,  wtBox + 1 * WSTRIDE, box_b + 256, actB, H, W, 256, 256, 256, 1, 0);
    conv(actB,  wtBox + 2 * WSTRIDE, box_b + 512, actA, H, W, 256, 256, 256, 1, 0);
    conv(actA,  wtBox + 3 * WSTRIDE, box_b + 768, actB, H, W, 256, 256, 256, 1, 0);
    conv(actB,  wtPred,  pred_b, deltas, H, W, 128, 16, 4, 0, 1);

    score_compact<<<(2 * HW * 80 + 255) / 256, 256, 0, stream>>>(
        logits, ctrO, HW, l, candV, candI, cnt);
    select_topk<<<2, 256, 0, stream>>>(candV, candI, cnt, l, selV, selI);
    decode_level<<<(2000 + 255) / 256, 256, 0, stream>>>(
        selV, selI, deltas, l, H, W, LS[l], catB, catS, catC);
  }

  nms_kernel<<<2, 256, 0, stream>>>(catB, catS, catC, (float*)d_out);
}
